// EpisodicGRU_74225624810000
// MI455X (gfx1250) — compile-verified, matrix operand order checked by reading, not run on hardware
//
#include <hip/hip_runtime.h>
#include <hip/hip_bf16.h>

// CDNA5 (gfx1250) wave32 WMMA types
typedef __attribute__((ext_vector_type(16))) __bf16 v16bf;
typedef __attribute__((ext_vector_type(8)))  float  v8f;

#define B_    64
#define T_    2048
#define I_    256
#define H_    256
#define G3H   768          // 3*H
#define ROWS  16           // batch rows per workgroup (WMMA M)
#define NT    48           // 768/16 N-tiles
#define KC    8            // 256/32 K-chunks
#define WAVES 16           // one wave per 16-wide H slice
#define THREADS (WAVES * 32)
#define WTILE_ELEMS (NT * KC * 32 * 16)   // bf16 elements per tiled weight

// ---------------------------------------------------------------------------
// Pre-tile W^T (K x N) into wave32 WMMA-B fragment layout, f32 -> bf16.
// Fragment (nt, kc): lane l element e holds B[k][n] with
//   n = nt*16 + (l%16),  k = kc*32 + 16*(l/16) + e     (contiguous k order, same as A)
// Flat: dst[((nt*KC + kc)*32 + lane)*16 + e] -> each lane reads 32 contiguous
// bytes at scan time (2x global_load_b128), fully coalesced, L2-resident.
// ---------------------------------------------------------------------------
__global__ void gru_prep_weights(const float* __restrict__ Wih,
                                 const float* __restrict__ Whh,
                                 __bf16* __restrict__ WihT,
                                 __bf16* __restrict__ WhhT) {
    int idx = blockIdx.x * blockDim.x + threadIdx.x;
    if (idx >= 2 * WTILE_ELEMS) return;
    const float* src = (idx < WTILE_ELEMS) ? Wih : Whh;
    __bf16*      dst = (idx < WTILE_ELEMS) ? WihT : WhhT;
    int i    = (idx < WTILE_ELEMS) ? idx : idx - WTILE_ELEMS;
    int e    = i & 15;
    int lane = (i >> 4) & 31;
    int kc   = (i >> 9) & 7;
    int nt   = i >> 12;
    int n = nt * 16 + (lane & 15);
    int k = kc * 32 + ((lane >> 4) << 4) + e;
    // (W^T)[k][n] == W[n][k], W is (768 x 256) row-major
    dst[i] = (__bf16)src[n * I_ + k];
}

// ---------------------------------------------------------------------------
// Fused episodic-GRU scan. One workgroup per 16 batch rows; 16 waves each own
// H columns [16w, 16w+16) for all three gates. Per step, per wave:
//   8 K-chunks x 6 WMMAs (gi_r/z/n from x-tile, gh_r/z/n from h-tile).
// Weight fragments are deliberately re-loaded from global each step (L2 hit):
// they cannot fit in registers (384 VGPRs/wave) or LDS (768 KB > 320 KB).
// The k-chunk order is rotated by t so the loads are not loop-invariant and
// the compiler cannot hoist-and-spill them to scratch.
// ---------------------------------------------------------------------------
__global__ __launch_bounds__(THREADS, 1)
void gru_scan(const float* __restrict__ x,        // (B,T,I)
              const float* __restrict__ attw,     // (B,T)
              const int*   __restrict__ lengths,  // (B,)
              const __bf16* __restrict__ WihT,    // tiled bf16 B-frags
              const __bf16* __restrict__ WhhT,    // tiled bf16 B-frags
              const float* __restrict__ b_ih,     // (768,)
              const float* __restrict__ b_hh,     // (768,)
              float* __restrict__ out) {          // (B,H)
    __shared__ __align__(32) __bf16 xb[ROWS * I_];   // x_t tile, WMMA-A bf16 (8 KB)
    __shared__ __align__(32) __bf16 hb[ROWS * H_];   // h tile,   WMMA-A bf16 (8 KB)
    __shared__ float hf[ROWS * H_];                  // h master copy f32   (16 KB)
    __shared__ float att_s[ROWS];
    __shared__ int   len_s[ROWS];

    const int tid     = threadIdx.x;
    const int lane    = tid & 31;
    const int wave    = tid >> 5;          // 0..15
    const int rowBase = blockIdx.x * ROWS;
    const int laneHi  = lane >> 4;         // 0 or 1 (wave32 half)

    // h0 = 0
    for (int i = tid; i < ROWS * H_; i += THREADS) {
        hf[i] = 0.0f;
        hb[i] = (__bf16)0.0f;
    }
    if (tid < ROWS) len_s[tid] = lengths[rowBase + tid];

    // Per-lane column in [0,256): C/D fragment col = lane%16 within wave slice.
    const int col = wave * 16 + (lane & 15);
    const float bir = b_ih[col], biz = b_ih[H_ + col], binn = b_ih[2 * H_ + col];
    const float bhr = b_hh[col], bhz = b_hh[H_ + col], bhn = b_hh[2 * H_ + col];

    // B-fragment bases for this wave's three gate nTiles (r: w, z: 16+w, n: 32+w)
    const __bf16* wir = WihT + ((size_t)((0  + wave) * KC) * 32 + lane) * 16;
    const __bf16* wiz = WihT + ((size_t)((16 + wave) * KC) * 32 + lane) * 16;
    const __bf16* win = WihT + ((size_t)((32 + wave) * KC) * 32 + lane) * 16;
    const __bf16* whr = WhhT + ((size_t)((0  + wave) * KC) * 32 + lane) * 16;
    const __bf16* whz = WhhT + ((size_t)((16 + wave) * KC) * 32 + lane) * 16;
    const __bf16* whn = WhhT + ((size_t)((32 + wave) * KC) * 32 + lane) * 16;
    const int fragStride = 32 * 16;        // bf16 elems per (nt,kc) fragment

    __syncthreads();

    for (int t = 0; t < T_; ++t) {
        // Stage x_t (16 rows x 256) -> bf16 WMMA-A tile. Coalesced over k.
        for (int i = tid; i < ROWS * I_; i += THREADS) {
            int r = i >> 8, k = i & (I_ - 1);
            xb[i] = (__bf16)x[((size_t)(rowBase + r) * T_ + t) * I_ + k];
        }
        if (tid < ROWS) att_s[tid] = attw[(size_t)(rowBase + tid) * T_ + t];
        __syncthreads();   // x tile + att ready; h tile from step t-1 visible

        // Prefetch next step's x tile while WMMAs run (global_prefetch_b8).
        if (t + 1 < T_) {
            int pr = tid >> 5, pk = (tid & 31) * 8;   // 16 rows x 32B granules
            __builtin_prefetch(&x[((size_t)(rowBase + pr) * T_ + (t + 1)) * I_ + pk], 0, 0);
        }

        v8f ir = {}, iz = {}, in_ = {}, hr = {}, hz = {}, hn = {};
        #pragma unroll
        for (int kc = 0; kc < KC; ++kc) {
            // Rotate k-chunk order by t: sum over k is commutative, but the
            // t-dependent addresses keep these as per-step global loads (L2
            // hits on the 768 KB tiled weights) instead of hoisted spills.
            const int kcr = (kc + t) & (KC - 1);
            // A-frags (16x32, contiguous k order as in B): row = lane%16,
            // K = kcr*32 + 16*(lane/16) + e  -> 32B contiguous per lane.
            const v16bf ax = *(const v16bf*)&xb[(lane & 15) * I_ + kcr * 32 + (laneHi << 4)];
            const v16bf ah = *(const v16bf*)&hb[(lane & 15) * H_ + kcr * 32 + (laneHi << 4)];
            const v16bf bir_f = *(const v16bf*)(wir + kcr * fragStride);
            const v16bf biz_f = *(const v16bf*)(wiz + kcr * fragStride);
            const v16bf bin_f = *(const v16bf*)(win + kcr * fragStride);
            const v16bf bhr_f = *(const v16bf*)(whr + kcr * fragStride);
            const v16bf bhz_f = *(const v16bf*)(whz + kcr * fragStride);
            const v16bf bhn_f = *(const v16bf*)(whn + kcr * fragStride);
            ir  = __builtin_amdgcn_wmma_f32_16x16x32_bf16(false, ax, false, bir_f, (short)0, ir,  false, false);
            iz  = __builtin_amdgcn_wmma_f32_16x16x32_bf16(false, ax, false, biz_f, (short)0, iz,  false, false);
            in_ = __builtin_amdgcn_wmma_f32_16x16x32_bf16(false, ax, false, bin_f, (short)0, in_, false, false);
            hr  = __builtin_amdgcn_wmma_f32_16x16x32_bf16(false, ah, false, bhr_f, (short)0, hr,  false, false);
            hz  = __builtin_amdgcn_wmma_f32_16x16x32_bf16(false, ah, false, bhz_f, (short)0, hz,  false, false);
            hn  = __builtin_amdgcn_wmma_f32_16x16x32_bf16(false, ah, false, bhn_f, (short)0, hn,  false, false);
        }

        // Gate math. C/D layout: element v of lane l -> row m = v + 8*(l/16),
        // col = lane%16 (within this wave's slice).
        float hnext[8];
        #pragma unroll
        for (int v = 0; v < 8; ++v) {
            const int   m    = v + (laneHi << 3);
            const float hold = hf[m * H_ + col];
            const float r  = 1.0f / (1.0f + __expf(-(ir[v] + bir + hr[v] + bhr)));
            const float z  = 1.0f / (1.0f + __expf(-(iz[v] + biz + hz[v] + bhz)));
            const float n  = tanhf(in_[v] + binn + r * (hn[v] + bhn));
            const float hnew = (1.0f - z) * n + z * hold;
            const float wt   = att_s[m];
            const float hg   = wt * hnew + (1.0f - wt) * hold;
            hnext[v] = (t < len_s[m]) ? hg : hold;
        }

        __syncthreads();   // all waves done reading hb/hf/xb for step t
        #pragma unroll
        for (int v = 0; v < 8; ++v) {
            const int m = v + (laneHi << 3);
            hf[m * H_ + col] = hnext[v];
            hb[m * H_ + col] = (__bf16)hnext[v];
        }
        // next iteration's first __syncthreads() publishes h to all waves
    }

    __syncthreads();
    for (int i = tid; i < ROWS * H_; i += THREADS) {
        int r = i >> 8, c = i & (H_ - 1);
        out[(size_t)(rowBase + r) * H_ + c] = hf[i];
    }
}

// ---------------------------------------------------------------------------
extern "C" void kernel_launch(void* const* d_in, const int* in_sizes, int n_in,
                              void* d_out, int out_size, void* d_ws, size_t ws_size,
                              hipStream_t stream) {
    const float* x    = (const float*)d_in[0];   // (B,T,I)
    const float* attw = (const float*)d_in[1];   // (B,T)
    const int*   len  = (const int*)  d_in[2];   // (B,)
    const float* Wih  = (const float*)d_in[3];   // (768,256)
    const float* Whh  = (const float*)d_in[4];   // (768,256)
    const float* bih  = (const float*)d_in[5];   // (768,)
    const float* bhh  = (const float*)d_in[6];   // (768,)
    float* out = (float*)d_out;                  // (B,H)

    __bf16* WihT = (__bf16*)d_ws;                // 384 KB
    __bf16* WhhT = WihT + WTILE_ELEMS;           // 384 KB  (ws total 768 KB)

    gru_prep_weights<<<(2 * WTILE_ELEMS + 255) / 256, 256, 0, stream>>>(Wih, Whh, WihT, WhhT);
    gru_scan<<<B_ / ROWS, THREADS, 0, stream>>>(x, attw, len, WihT, WhhT, bih, bhh, out);
}
